// SRUpp_1906965479748
// MI455X (gfx1250) — compile-verified, matrix operand order checked by reading, not run on hardware
//
#include <hip/hip_runtime.h>
#include <hip/hip_bf16.h>
#include <stdint.h>

// ---------------- problem constants ----------------
#define BB   8
#define LL   2048
#define DD   1024
#define AA   256
#define HH3  (3 * DD)

typedef __attribute__((ext_vector_type(16))) __bf16 v16bf;
typedef __attribute__((ext_vector_type(8)))  float  v8f;

union Frag { v16bf v; uint32_t u[8]; };

// ---------------- helpers ----------------
__device__ __forceinline__ uint16_t f32_to_bf16_bits(float f) {
    uint32_t a = __float_as_uint(f);
    return (uint16_t)((a + 0x7FFFu + ((a >> 16) & 1u)) >> 16);
}
__device__ __forceinline__ uint32_t f32_pair_to_bf16(float lo, float hi) {
    uint32_t a = __float_as_uint(lo), b = __float_as_uint(hi);
    a = (a + 0x7FFFu + ((a >> 16) & 1u)) >> 16;
    b = (b + 0x7FFFu + ((b >> 16) & 1u)) >> 16;
    return (b << 16) | (a & 0xFFFFu);
}

// Fragment load for V_WMMA_F32_16X16X32_BF16 (wave32).
// 16-bit A-matrix 16x32 layout (ISA 7.12.2): lane = m + 16*half,
// VGPR v holds the k-pair at ((v>>2)<<4) | (half<<3) | ((v&3)<<1).
// Consecutive v are contiguous in memory -> lowers to 2x global_load_b128.
// B fragments (column n of B with K contiguous in a transposed array) use
// the identical layout with the 16-dim being N.
__device__ __forceinline__ v16bf load_frag_bf16(const uint16_t* __restrict__ base,
                                                int ld, int row0, int k0, int lane) {
    const int m = lane & 15, g = lane >> 4;
    Frag f;
#pragma unroll
    for (int r = 0; r < 8; ++r) {
        const int kpb = ((r >> 2) << 4) | (g << 3) | ((r & 3) << 1);
        f.u[r] = *(const uint32_t*)(base + (size_t)(row0 + m) * ld + k0 + kpb);
    }
    return f.v;
}

__device__ __forceinline__ v8f wmma_bf16(v16bf a, v16bf b, v8f c) {
#if defined(__HIP_DEVICE_COMPILE__)
    return __builtin_amdgcn_wmma_f32_16x16x32_bf16(false, a, false, b, (short)0, c,
                                                   false, false);
#else
    return c;
#endif
}

#define ZERO8 {0.f,0.f,0.f,0.f,0.f,0.f,0.f,0.f}

// ---------------- kernel 0a: weight convert + transpose to bf16 ----------------
__global__ void __launch_bounds__(256)
wconv_kernel(const float* __restrict__ Wq, const float* __restrict__ Wk,
             const float* __restrict__ Wv, const float* __restrict__ Wu,
             uint16_t* __restrict__ Wqt, uint16_t* __restrict__ Wkt,
             uint16_t* __restrict__ Wvt, uint16_t* __restrict__ Wut) {
    const int NW = DD * AA;
    int i = blockIdx.x * 256 + threadIdx.x;
    if (i < 3 * NW) {
        int w = i / NW, rem = i % NW;
        int d = rem / AA, a = rem % AA;
        const float* src = (w == 0) ? Wq : (w == 1) ? Wk : Wv;
        uint16_t* dst = (w == 0) ? Wqt : (w == 1) ? Wkt : Wvt;
        dst[(size_t)a * DD + d] = f32_to_bf16_bits(src[rem]);     // [A][D]
    } else {
        int j = i - 3 * NW;
        if (j < AA * HH3) {
            int a = j / HH3, h = j % HH3;
            Wut[(size_t)h * AA + a] = f32_to_bf16_bits(Wu[j]);    // [3D][A]
        }
    }
}

// ---------------- kernel 0b: x f32 -> bf16 (one-time) ----------------
__global__ void __launch_bounds__(256)
xconv_kernel(const float* __restrict__ x, uint16_t* __restrict__ xb) {
    const size_t i = ((size_t)blockIdx.x * 256 + threadIdx.x) * 2;
    float2 f = *(const float2*)(x + i);
    *(uint32_t*)(xb + i) = f32_pair_to_bf16(f.x, f.y);
}

// ---------------- kernel 1: Q/K/V projection (WMMA) ----------------
// grid = (32, B, 3), block = 256 (8 waves). Each wave: one 32x64 output tile.
__global__ void __launch_bounds__(256)
qkv_kernel(const uint16_t* __restrict__ xb,
           const uint16_t* __restrict__ Wqt, const uint16_t* __restrict__ Wkt,
           const uint16_t* __restrict__ Wvt,
           float* __restrict__ Qf, uint16_t* __restrict__ Qb,
           uint16_t* __restrict__ Kb, uint16_t* __restrict__ Vt) {
    const int lane = threadIdx.x & 31;
    const int wid  = (blockIdx.x << 3) | (threadIdx.x >> 5);   // 0..255
    const int b = blockIdx.y, which = blockIdx.z;
    const int m0 = (wid >> 2) * 32;
    const int n0 = (wid & 3) * 64;
    const uint16_t* Wt = (which == 0) ? Wqt : (which == 1) ? Wkt : Wvt;
    const uint16_t* xbb = xb + (size_t)b * LL * DD;

    v8f acc[2][4] = {{ZERO8, ZERO8, ZERO8, ZERO8}, {ZERO8, ZERO8, ZERO8, ZERO8}};
    for (int k0 = 0; k0 < DD; k0 += 32) {
        v16bf a0 = load_frag_bf16(xbb, DD, m0,      k0, lane);
        v16bf a1 = load_frag_bf16(xbb, DD, m0 + 16, k0, lane);
#pragma unroll
        for (int j = 0; j < 4; ++j) {
            v16bf bfr = load_frag_bf16(Wt, DD, n0 + j * 16, k0, lane);
            acc[0][j] = wmma_bf16(a0, bfr, acc[0][j]);
            acc[1][j] = wmma_bf16(a1, bfr, acc[1][j]);
        }
    }
    const int n_lo = lane & 15, rb = (lane >> 4) << 3;
    if (which == 0) {
#pragma unroll
        for (int i = 0; i < 2; ++i)
#pragma unroll
            for (int j = 0; j < 4; ++j) {
                const int n = n0 + j * 16 + n_lo;
#pragma unroll
                for (int r = 0; r < 8; ++r) {
                    const int mm = m0 + i * 16 + rb + r;
                    const float val = acc[i][j][r];
                    Qf[((size_t)b * LL + mm) * AA + n] = val;
                    Qb[((size_t)b * LL + mm) * AA + n] = f32_to_bf16_bits(val);
                }
            }
    } else if (which == 1) {
#pragma unroll
        for (int i = 0; i < 2; ++i)
#pragma unroll
            for (int j = 0; j < 4; ++j) {
                const int n = n0 + j * 16 + n_lo;
#pragma unroll
                for (int r = 0; r < 8; ++r)
                    Kb[((size_t)b * LL + (m0 + i * 16 + rb + r)) * AA + n] =
                        f32_to_bf16_bits(acc[i][j][r]);
            }
    } else {                                            // V stored transposed [A][L]
#pragma unroll
        for (int i = 0; i < 2; ++i)
#pragma unroll
            for (int j = 0; j < 4; ++j) {
                const int n = n0 + j * 16 + n_lo;
#pragma unroll
                for (int r = 0; r < 8; ++r)
                    Vt[((size_t)b * AA + n) * LL + (m0 + i * 16 + rb + r)] =
                        f32_to_bf16_bits(acc[i][j][r]);
            }
    }
}

// ---------------- kernel 2: S = Q K^T / sqrt(A) (WMMA) ----------------
// grid = (256, B), block = 256. wave tile 32x64, K-loop over A=256.
__global__ void __launch_bounds__(256)
scores_kernel(const uint16_t* __restrict__ Qb, const uint16_t* __restrict__ Kb,
              float* __restrict__ S) {
    const int lane = threadIdx.x & 31;
    const int wid  = (blockIdx.x << 3) | (threadIdx.x >> 5);   // 0..2047
    const int b = blockIdx.y;
    const int m0 = (wid >> 5) * 32;
    const int n0 = (wid & 31) * 64;
    const uint16_t* qb = Qb + (size_t)b * LL * AA;
    const uint16_t* kb = Kb + (size_t)b * LL * AA;

    v8f acc[2][4] = {{ZERO8, ZERO8, ZERO8, ZERO8}, {ZERO8, ZERO8, ZERO8, ZERO8}};
    for (int k0 = 0; k0 < AA; k0 += 32) {
        v16bf a0 = load_frag_bf16(qb, AA, m0,      k0, lane);
        v16bf a1 = load_frag_bf16(qb, AA, m0 + 16, k0, lane);
#pragma unroll
        for (int j = 0; j < 4; ++j) {
            v16bf bfr = load_frag_bf16(kb, AA, n0 + j * 16, k0, lane);
            acc[0][j] = wmma_bf16(a0, bfr, acc[0][j]);
            acc[1][j] = wmma_bf16(a1, bfr, acc[1][j]);
        }
    }
    float* sb = S + (size_t)b * LL * LL;
    const int n_lo = lane & 15, rb = (lane >> 4) << 3;
#pragma unroll
    for (int i = 0; i < 2; ++i)
#pragma unroll
        for (int j = 0; j < 4; ++j) {
            const int n = n0 + j * 16 + n_lo;
#pragma unroll
            for (int r = 0; r < 8; ++r)
                sb[(size_t)(m0 + i * 16 + rb + r) * LL + n] =
                    acc[i][j][r] * 0.0625f;                    // 1/sqrt(256)
        }
}

// ---------------- kernel 3: row softmax, f32 -> bf16 probs ----------------
// grid = (L, B), block = 256, 8 elems/thread.
__global__ void __launch_bounds__(256)
softmax_kernel(const float* __restrict__ S, uint16_t* __restrict__ P) {
    __shared__ float red[256];
    const int row = blockIdx.x, b = blockIdx.y, t = threadIdx.x;
    const float* s = S + ((size_t)b * LL + row) * LL;
    uint16_t*    p = P + ((size_t)b * LL + row) * LL;
    float v[8], m = -3.4e38f;
#pragma unroll
    for (int i = 0; i < 8; ++i) { v[i] = s[t + 256 * i]; m = fmaxf(m, v[i]); }
    red[t] = m; __syncthreads();
    for (int off = 128; off > 0; off >>= 1) {
        if (t < off) red[t] = fmaxf(red[t], red[t + off]);
        __syncthreads();
    }
    m = red[0]; __syncthreads();
    float sum = 0.f;
#pragma unroll
    for (int i = 0; i < 8; ++i) { v[i] = __expf(v[i] - m); sum += v[i]; }
    red[t] = sum; __syncthreads();
    for (int off = 128; off > 0; off >>= 1) {
        if (t < off) red[t] += red[t + off];
        __syncthreads();
    }
    const float inv = 1.0f / red[0];
#pragma unroll
    for (int i = 0; i < 8; ++i) p[t + 256 * i] = f32_to_bf16_bits(v[i] * inv);
}

// ---------------- kernel 4: F = P V, fused mixed = alpha*F + Q -> bf16 ----------------
// grid = (32, B), block = 256. wave tile 32x64, K-loop over L=2048.
__global__ void __launch_bounds__(256)
pv_mixed_kernel(const uint16_t* __restrict__ P, const uint16_t* __restrict__ Vt,
                const float* __restrict__ Qf, const float* __restrict__ alphaPtr,
                uint16_t* __restrict__ Mixed) {
    const int lane = threadIdx.x & 31;
    const int wid  = (blockIdx.x << 3) | (threadIdx.x >> 5);   // 0..255
    const int b = blockIdx.y;
    const int m0 = (wid >> 2) * 32;
    const int n0 = (wid & 3) * 64;
    const uint16_t* pb = P  + (size_t)b * LL * LL;
    const uint16_t* vb = Vt + (size_t)b * AA * LL;

    v8f acc[2][4] = {{ZERO8, ZERO8, ZERO8, ZERO8}, {ZERO8, ZERO8, ZERO8, ZERO8}};
    for (int k0 = 0; k0 < LL; k0 += 32) {
        v16bf a0 = load_frag_bf16(pb, LL, m0,      k0, lane);
        v16bf a1 = load_frag_bf16(pb, LL, m0 + 16, k0, lane);
#pragma unroll
        for (int j = 0; j < 4; ++j) {
            v16bf bfr = load_frag_bf16(vb, LL, n0 + j * 16, k0, lane);
            acc[0][j] = wmma_bf16(a0, bfr, acc[0][j]);
            acc[1][j] = wmma_bf16(a1, bfr, acc[1][j]);
        }
    }
    const float alpha = alphaPtr[0];
    const int n_lo = lane & 15, rb = (lane >> 4) << 3;
#pragma unroll
    for (int i = 0; i < 2; ++i)
#pragma unroll
        for (int j = 0; j < 4; ++j) {
            const int n = n0 + j * 16 + n_lo;
#pragma unroll
            for (int r = 0; r < 8; ++r) {
                const size_t o = ((size_t)b * LL + (m0 + i * 16 + rb + r)) * AA + n;
                Mixed[o] = f32_to_bf16_bits(alpha * acc[i][j][r] + Qf[o]);
            }
        }
}

// ---------------- kernel 5: U = Mixed @ Wu (WMMA) ----------------
// grid = (384, B), block = 256. M=2048 (64 blocks of 32), N=3072 (48 groups of 64).
__global__ void __launch_bounds__(256)
u_gemm_kernel(const uint16_t* __restrict__ Mixed, const uint16_t* __restrict__ Wut,
              float* __restrict__ U) {
    const int lane = threadIdx.x & 31;
    const int wid  = (blockIdx.x << 3) | (threadIdx.x >> 5);   // 0..3071
    const int b = blockIdx.y;
    const int m0 = (wid / 48) * 32;
    const int n0 = (wid % 48) * 64;
    const uint16_t* mb = Mixed + (size_t)b * LL * AA;

    v8f acc[2][4] = {{ZERO8, ZERO8, ZERO8, ZERO8}, {ZERO8, ZERO8, ZERO8, ZERO8}};
    for (int k0 = 0; k0 < AA; k0 += 32) {
        v16bf a0 = load_frag_bf16(mb, AA, m0,      k0, lane);
        v16bf a1 = load_frag_bf16(mb, AA, m0 + 16, k0, lane);
#pragma unroll
        for (int j = 0; j < 4; ++j) {
            v16bf bfr = load_frag_bf16(Wut, AA, n0 + j * 16, k0, lane);
            acc[0][j] = wmma_bf16(a0, bfr, acc[0][j]);
            acc[1][j] = wmma_bf16(a1, bfr, acc[1][j]);
        }
    }
    float* ub = U + (size_t)b * LL * HH3;
    const int n_lo = lane & 15, rb = (lane >> 4) << 3;
#pragma unroll
    for (int i = 0; i < 2; ++i)
#pragma unroll
        for (int j = 0; j < 4; ++j) {
            const int n = n0 + j * 16 + n_lo;
#pragma unroll
            for (int r = 0; r < 8; ++r)
                ub[(size_t)(m0 + i * 16 + rb + r) * HH3 + n] = acc[i][j][r];
        }
}

// ---------------- kernel 6: SRU recurrence ----------------
// One thread per (b, d) channel; sequential over t with software prefetch.
__global__ void __launch_bounds__(256)
sru_kernel(const float* __restrict__ U, const float* __restrict__ x,
           const float* __restrict__ vf, const float* __restrict__ vr,
           const float* __restrict__ bf, const float* __restrict__ br,
           float* __restrict__ out) {
    const int idx = blockIdx.x * 256 + threadIdx.x;   // 0..B*D-1
    const int b = idx / DD, d = idx % DD;
    const float* Ub = U + (size_t)b * LL * HH3;
    const float* xb = x + (size_t)b * LL * DD;
    float* ob = out + (size_t)b * LL * DD;
    const float vfd = vf[d], vrd = vr[d], bfd = bf[d], brd = br[d];
    float c = 0.f;
    for (int t = 0; t < LL; ++t) {
        if (t + 4 < LL) {
            __builtin_prefetch(&Ub[(size_t)(t + 4) * HH3 + d], 0, 0);
            __builtin_prefetch(&xb[(size_t)(t + 4) * DD + d], 0, 0);
        }
        const float uf = Ub[(size_t)t * HH3 + d];
        const float ur = Ub[(size_t)t * HH3 + DD + d];
        const float uh = Ub[(size_t)t * HH3 + 2 * DD + d];
        const float xv = xb[(size_t)t * DD + d];
        const float f = 1.f / (1.f + __expf(-(uf + vfd * c + bfd)));
        const float r = 1.f / (1.f + __expf(-(ur + vrd * c + brd)));  // uses c_prev
        c = f * c + (1.f - f) * uh;
        ob[(size_t)t * DD + d] = r * c + (1.f - r) * xv;
    }
}

// ---------------- launcher ----------------
extern "C" void kernel_launch(void* const* d_in, const int* in_sizes, int n_in,
                              void* d_out, int out_size, void* d_ws, size_t ws_size,
                              hipStream_t stream) {
    (void)in_sizes; (void)n_in; (void)out_size; (void)ws_size;
    const float* x     = (const float*)d_in[0];
    const float* Wq    = (const float*)d_in[1];
    const float* Wk    = (const float*)d_in[2];
    const float* Wv    = (const float*)d_in[3];
    const float* Wu    = (const float*)d_in[4];
    const float* v_f   = (const float*)d_in[5];
    const float* v_r   = (const float*)d_in[6];
    const float* b_f   = (const float*)d_in[7];
    const float* b_r   = (const float*)d_in[8];
    const float* alpha = (const float*)d_in[9];
    float* out = (float*)d_out;

    // workspace layout (bytes)
    char* ws = (char*)d_ws;
    uint16_t* Wqt   = (uint16_t*)(ws + 0);                       // 512 KB
    uint16_t* Wkt   = (uint16_t*)(ws + (512u << 10));            // 512 KB
    uint16_t* Wvt   = (uint16_t*)(ws + (1024u << 10));           // 512 KB
    uint16_t* Wut   = (uint16_t*)(ws + (1536u << 10));           // 1.5 MB
    char* p = ws + 3145728;
    float*    Qf    = (float*)p;                p += (size_t)BB * LL * AA * 4;   // 16 MB
    uint16_t* Qb    = (uint16_t*)p;             p += (size_t)BB * LL * AA * 2;   // 8 MB
    uint16_t* Kb    = (uint16_t*)p;             p += (size_t)BB * LL * AA * 2;   // 8 MB
    uint16_t* Vt    = (uint16_t*)p;             p += (size_t)BB * LL * AA * 2;   // 8 MB
    uint16_t* Mixed = (uint16_t*)p;             p += (size_t)BB * LL * AA * 2;   // 8 MB
    // big region (192 MB), time-multiplexed:
    //   phase 0 (QKV GEMM):   xb bf16 (32 MB)
    //   phase 1 (attention):  S f32 (128 MB) + P bf16 (64 MB)
    //   phase 2 (U + SRU):    U f32 (192 MB)
    char* big = p;
    uint16_t* xbp = (uint16_t*)big;                              // 32 MB
    float*    S   = (float*)big;                                 // 128 MB
    uint16_t* P   = (uint16_t*)(big + (size_t)BB * LL * LL * 4); // 64 MB
    float*    U   = (float*)big;                                 // 192 MB (reuse)

    wconv_kernel<<<dim3(6144), dim3(256), 0, stream>>>(Wq, Wk, Wv, Wu,
                                                       Wqt, Wkt, Wvt, Wut);
    xconv_kernel<<<dim3((BB * LL * DD) / 512), dim3(256), 0, stream>>>(x, xbp);
    qkv_kernel<<<dim3(32, BB, 3), dim3(256), 0, stream>>>(xbp, Wqt, Wkt, Wvt,
                                                          Qf, Qb, Kb, Vt);
    scores_kernel<<<dim3(256, BB), dim3(256), 0, stream>>>(Qb, Kb, S);
    softmax_kernel<<<dim3(LL, BB), dim3(256), 0, stream>>>(S, P);
    pv_mixed_kernel<<<dim3(32, BB), dim3(256), 0, stream>>>(P, Vt, Qf, alpha, Mixed);
    u_gemm_kernel<<<dim3(384, BB), dim3(256), 0, stream>>>(Mixed, Wut, U);
    sru_kernel<<<dim3((BB * DD) / 256), dim3(256), 0, stream>>>(U, x, v_f, v_r,
                                                                b_f, b_r, out);
}
